// TemporalGRU_61443802136844
// MI455X (gfx1250) — compile-verified, matrix operand order checked by reading, not run on hardware
//
#include <hip/hip_runtime.h>

// ---------------- types ----------------
typedef __attribute__((ext_vector_type(16))) __bf16 bf16x16;
typedef __attribute__((ext_vector_type(8)))  __bf16 bf16x8;
typedef __attribute__((ext_vector_type(8)))  float  f32x8;
typedef __attribute__((ext_vector_type(4)))  float  f32x4;

// ---------------- problem constants ----------------
#define IN_DIM   64
#define HIDDEN   256
#define GATES3   (3 * HIDDEN)      // 768
#define BATCH    32
#define SEQLEN   24
#define NODES    207
#define BN       (BATCH * NODES)   // 6624 independent sequences
#define JTILES   (HIDDEN / 16)     // 16 column tiles
#define WAVES    4                 // waves per workgroup
#define LDS_LD   264               // bf16 elems per h row (256 + 8 pad; 16B-aligned rows)
#define FRAG     512               // bf16 elems per packed 16x32 B-fragment (1 KB)

// ---------------- helpers ----------------
__device__ __forceinline__ float sigmoidf_fast(float x) {
    return 1.0f / (1.0f + __expf(-x));
}
__device__ __forceinline__ float tanhf_fast(float x) {
    return 1.0f - 2.0f / (__expf(2.0f * x) + 1.0f);
}

// A-fragment (16x32 bf16, ISA 7.12.2 layout) from an fp32 row.
// kb already includes the per-lane half offset (lanes 0-15: +0, lanes 16-31: +8).
// halves 0..7 <- K = kb+0..7 ; halves 8..15 <- K = kb+16..23
__device__ __forceinline__ bf16x16 load_a_frag_f32(const float* __restrict__ rowp, int kb) {
    f32x4 v0 = *(const f32x4*)(rowp + kb);
    f32x4 v1 = *(const f32x4*)(rowp + kb + 4);
    f32x4 v2 = *(const f32x4*)(rowp + kb + 16);
    f32x4 v3 = *(const f32x4*)(rowp + kb + 20);
    bf16x16 a;
#pragma unroll
    for (int i = 0; i < 4; ++i) {
        a[i]      = (__bf16)v0[i];
        a[i + 4]  = (__bf16)v1[i];
        a[i + 8]  = (__bf16)v2[i];
        a[i + 12] = (__bf16)v3[i];
    }
    return a;
}

// Fragment from the bf16 LDS h-mirror (strided row layout: hi chunk at +16).
__device__ __forceinline__ bf16x16 load_frag_lds(const __bf16* p) {
    bf16x8 lo = *(const bf16x8*)(p);
    bf16x8 hi = *(const bf16x8*)(p + 16);
    return __builtin_shufflevector(lo, hi, 0, 1, 2, 3, 4, 5, 6, 7,
                                           8, 9, 10, 11, 12, 13, 14, 15);
}

// Fragment from packed weights: p = block + 16B*lane; hi chunk lives at +512B.
// Both wave-level loads are fully coalesced 512B bursts.
__device__ __forceinline__ bf16x16 load_frag_pack(const __bf16* p) {
    bf16x8 lo = *(const bf16x8*)(p);
    bf16x8 hi = *(const bf16x8*)(p + 256);
    return __builtin_shufflevector(lo, hi, 0, 1, 2, 3, 4, 5, 6, 7,
                                           8, 9, 10, 11, 12, 13, 14, 15);
}

__device__ __forceinline__ f32x8 wmma_bf16(bf16x16 a, bf16x16 b, f32x8 c) {
    // (neg_a, A, neg_b, B, c_mod, C, reuse_a, reuse_b)
    return __builtin_amdgcn_wmma_f32_16x16x32_bf16(false, a, false, b,
                                                   (short)0, c, false, false);
}

// ------------- weight pack (fp32 -> bf16, fragment-major), once per call -------
// Fragment id: W_ih -> (j*2 + kc)*3 + g   (kc in [0,2))
//              W_hh -> (j*8 + kc)*3 + g   (kc in [0,8))
// Block of 512 bf16: [lo: lane*8+e] [hi at +256: lane*8+e]
//   lane n = lane&15 -> W row g*256 + j*16 + n
//   K      = kc*32 + (lane>>4)*8 + hi*16 + e
__global__ void pack_weights_kernel(const float* __restrict__ Wih_f,
                                    const float* __restrict__ Whh_f,
                                    __bf16* __restrict__ Wih_p,
                                    __bf16* __restrict__ Whh_p) {
    const int d = blockIdx.x * blockDim.x + threadIdx.x;
    if (d < GATES3 * IN_DIM) {
        const int frag = d >> 9, w = d & 511;
        const int hi = w >> 8, rem = w & 255;
        const int lane = rem >> 3, e = rem & 7;
        const int g = frag % 3, jkc = frag / 3;
        const int kc = jkc & 1, j = jkc >> 1;
        const int srow = g * HIDDEN + j * 16 + (lane & 15);
        const int scol = kc * 32 + ((lane >> 4) << 3) + hi * 16 + e;
        Wih_p[d] = (__bf16)Wih_f[srow * IN_DIM + scol];
    }
    if (d < GATES3 * HIDDEN) {
        const int frag = d >> 9, w = d & 511;
        const int hi = w >> 8, rem = w & 255;
        const int lane = rem >> 3, e = rem & 7;
        const int g = frag % 3, jkc = frag / 3;
        const int kc = jkc & 7, j = jkc >> 3;
        const int srow = g * HIDDEN + j * 16 + (lane & 15);
        const int scol = kc * 32 + ((lane >> 4) << 3) + hi * 16 + e;
        Whh_p[d] = (__bf16)Whh_f[srow * HIDDEN + scol];
    }
}

// ---------------- persistent GRU kernel: all 24 timesteps in one launch ----------
__global__ __launch_bounds__(WAVES * 32)
void gru_persistent_kernel(const float* __restrict__ x_seq,   // (B, L, N, F) fp32
                           const __bf16* __restrict__ Wih,    // packed fragments
                           const __bf16* __restrict__ Whh,    // packed fragments
                           const float* __restrict__ b_ih,    // (768)
                           const float* __restrict__ b_hh,    // (768)
                           float* __restrict__ out) {         // (BN, 256) fp32
    __shared__ __align__(16) __bf16 hs[WAVES][16][LDS_LD];

    const int lane    = threadIdx.x & 31;
    const int wave    = threadIdx.x >> 5;
    const int rowbase = (blockIdx.x * WAVES + wave) * 16;
    if (rowbase >= BN) return;                 // uniform per wave: EXEC stays all-1s

    const int m     = lane & 15;               // A row-in-tile / C col-in-tile
    const int halfk = (lane >> 4) << 3;        // 0 or 8 (16-bit fragment half split)
    const int rtop  = (lane >> 4) << 3;        // C/D: vgpr i -> local row rtop+i
    const int row   = rowbase + m;

    const int    bidx  = row / NODES;
    const int    node  = row - bidx * NODES;
    const float* xbase = x_seq + ((size_t)bidx * SEQLEN * NODES + node) * IN_DIM;
    const size_t xstep = (size_t)NODES * IN_DIM;   // stride between timesteps

#pragma unroll 1
    for (int t = 0; t < SEQLEN; ++t) {
        // ---- A fragments for x_t (global fp32 -> bf16), prefetch x_{t+1}
        const float* xrow = xbase + (size_t)t * xstep;
        bf16x16 ax0 = load_a_frag_f32(xrow, 0 + halfk);
        bf16x16 ax1 = load_a_frag_f32(xrow, 32 + halfk);
        if (t + 1 < SEQLEN) __builtin_prefetch(xrow + xstep + halfk, 0, 1);

        // ---- A fragments for h_t from the bf16 LDS mirror (ds_load_b128)
        bf16x16 ah[8];
        if (t > 0) {
#pragma unroll
            for (int kc = 0; kc < 8; ++kc)
                ah[kc] = load_frag_lds(&hs[wave][m][kc * 32 + halfk]);
        }

        // ---- 16 column tiles; ROLLED so live ranges stay small (no spills)
#pragma unroll 1
        for (int j = 0; j < JTILES; ++j) {
            const int cb = j * 16;
            // packed fragment bases: all loads below are immediate offsets
            const __bf16* pih = Wih + (size_t)j * (6 * FRAG) + (lane << 3);
            const __bf16* phh = Whh + (size_t)j * (24 * FRAG) + (lane << 3);
            // warm next j-tile's 24KB weight block (lanes fan across it)
            if (j + 1 < JTILES)
                __builtin_prefetch(Whh + (size_t)(j + 1) * (24 * FRAG) + lane * 384, 0, 1);

            f32x8 acc_r = {}, acc_z = {}, acc_in = {}, acc_hn = {};

            // input-side GEMM: x_t @ W_ih^T  (K = 64)
            acc_r  = wmma_bf16(ax0, load_frag_pack(pih + 0 * FRAG), acc_r);
            acc_z  = wmma_bf16(ax0, load_frag_pack(pih + 1 * FRAG), acc_z);
            acc_in = wmma_bf16(ax0, load_frag_pack(pih + 2 * FRAG), acc_in);
            acc_r  = wmma_bf16(ax1, load_frag_pack(pih + 3 * FRAG), acc_r);
            acc_z  = wmma_bf16(ax1, load_frag_pack(pih + 4 * FRAG), acc_z);
            acc_in = wmma_bf16(ax1, load_frag_pack(pih + 5 * FRAG), acc_in);

            // recurrent GEMM: h_t @ W_hh^T  (K = 256)
            if (t > 0) {
#pragma unroll
                for (int kc = 0; kc < 8; ++kc) {
                    acc_r  = wmma_bf16(ah[kc], load_frag_pack(phh + (kc * 3 + 0) * FRAG), acc_r);
                    acc_z  = wmma_bf16(ah[kc], load_frag_pack(phh + (kc * 3 + 1) * FRAG), acc_z);
                    acc_hn = wmma_bf16(ah[kc], load_frag_pack(phh + (kc * 3 + 2) * FRAG), acc_hn);
                }
            }

            // ---- gates + state update (C layout: col = cb+m, local row = rtop+i)
            const int   col     = cb + m;
            const float bias_r  = b_ih[col] + b_hh[col];              // L0-resident
            const float bias_z  = b_ih[HIDDEN + col] + b_hh[HIDDEN + col];
            const float bias_in = b_ih[2 * HIDDEN + col];
            const float bias_hn = b_hh[2 * HIDDEN + col];

#pragma unroll
            for (int i = 0; i < 8; ++i) {
                const float h_old = (t > 0) ? (float)hs[wave][rtop + i][col] : 0.0f;
                const float rg    = sigmoidf_fast(acc_r[i] + bias_r);
                const float zg    = sigmoidf_fast(acc_z[i] + bias_z);
                const float ghn   = acc_hn[i] + bias_hn;   // t==0: acc_hn==0 -> b_hh
                const float ng    = tanhf_fast(acc_in[i] + bias_in + rg * ghn);
                const float hnew  = (1.0f - zg) * ng + zg * h_old;
                if (t == SEQLEN - 1) {
                    out[(size_t)(rowbase + rtop + i) * HIDDEN + col] = hnew;
                } else {
                    hs[wave][rtop + i][col] = (__bf16)hnew;  // state for next step
                }
            }
        }
    }
}

// ---------------- host entry ----------------
extern "C" void kernel_launch(void* const* d_in, const int* in_sizes, int n_in,
                              void* d_out, int out_size, void* d_ws, size_t ws_size,
                              hipStream_t stream) {
    const float* x_seq = (const float*)d_in[0];
    const float* W_ih  = (const float*)d_in[1];
    const float* W_hh  = (const float*)d_in[2];
    const float* b_ih  = (const float*)d_in[3];
    const float* b_hh  = (const float*)d_in[4];
    float*       out   = (float*)d_out;

    // workspace layout: [ W_ih packed bf16 | W_hh packed bf16 ]
    __bf16* Wih_p = (__bf16*)d_ws;
    __bf16* Whh_p = Wih_p + (size_t)GATES3 * IN_DIM;

    pack_weights_kernel<<<(GATES3 * HIDDEN + 255) / 256, 256, 0, stream>>>(
        W_ih, W_hh, Wih_p, Whh_p);

    const int grid = (BN + WAVES * 16 - 1) / (WAVES * 16);  // 104 workgroups
    gru_persistent_kernel<<<grid, WAVES * 32, 0, stream>>>(x_seq, Wih_p, Whh_p,
                                                           b_ih, b_hh, out);
}
